// Flow_76845554860454
// MI455X (gfx1250) — matrix operand order repaired and checked by reading, compiled, not run on hardware
//
#include <hip/hip_runtime.h>
#include <hip/hip_fp16.h>

// ---------------------------------------------------------------------------
// Types
// ---------------------------------------------------------------------------
typedef __attribute__((ext_vector_type(16))) _Float16 v16h;
typedef __attribute__((ext_vector_type(2)))  _Float16 v2h;
typedef __attribute__((ext_vector_type(8)))  float    v8f;
typedef __attribute__((ext_vector_type(8)))  int      v8i;

#define B_  256
#define C_  64
#define G_  2048
#define H_  256

#define LOG2E  1.44269504089f
#define HLOG2E 0.721347520444f        // 0.5 * log2(e)

// pack two f32 -> f16 pair in one VGPR (v_cvt_pk_rtz_f16_f32)
__device__ __forceinline__ v2h pk2(float a, float b) {
    return __builtin_bit_cast(v2h, __builtin_amdgcn_cvt_pkrtz(a, b));
}

__device__ __forceinline__ unsigned short f2h(float f) {
    _Float16 h = (_Float16)f;
    return __builtin_bit_cast(unsigned short, h);
}

// fence: same-wave cross-lane LDS handoff (in-order DS pipe + compiler barrier)
__device__ __forceinline__ void lds_fence_wave() {
    __builtin_amdgcn_wave_barrier();
    asm volatile("s_wait_dscnt 0" ::: "memory");
    __builtin_amdgcn_wave_barrier();
}

__device__ __forceinline__ v8f wmma_f16(v16h a, v16h b, v8f c) {
    return __builtin_amdgcn_wmma_f32_16x16x32_f16(
        /*neg_a=*/false, a, /*neg_b=*/false, b,
        /*c_mod=*/(short)0, c, /*reuse_a=*/false, /*reuse_b=*/false);
}

// ---------------------------------------------------------------------------
// Cheap transcendentals: prefer hardware v_tanh_f32 (CDNA5 TRANS op)
// ---------------------------------------------------------------------------
__device__ __forceinline__ float fast_exp(float x) {
    return __builtin_amdgcn_exp2f(x * LOG2E);
}

#if __has_builtin(__builtin_amdgcn_tanhf)
#define HAVE_TANH 1
#else
#define HAVE_TANH 0
#endif

#if HAVE_TANH
// tanh-form GELU: max |err| vs exact erf-GELU ~3e-4 (below f16 noise floor)
__device__ __forceinline__ v8f gelu8(v8f x) {
    v8f x2 = x * x;
    v8f inner = x * (x2 * 0.03567740814f + 0.7978845608f);
    v8f th;
#pragma unroll
    for (int r = 0; r < 8; ++r) th[r] = __builtin_amdgcn_tanhf(inner[r]);
    v8f hx = x * 0.5f;
    return hx * th + hx;
}
#else
__device__ __forceinline__ float fast_sigmoid(float x) {
    return __builtin_amdgcn_rcpf(1.0f + __builtin_amdgcn_exp2f(x * -LOG2E));
}
// Exact-GELU via Abramowitz–Stegun 7.1.26 erf (|err| <= 1.5e-7).
__device__ __forceinline__ v8f gelu8(v8f x) {
    const float A1 = 0.254829592f, A2 = -0.284496736f, A3 = 1.421413741f,
                A4 = -1.453152027f, A5 = 1.061405429f, P = 0.3275911f;
    v8f z   = __builtin_elementwise_abs(x) * 0.70710678118f;   // |x|/sqrt(2)
    v8f den = z * P + 1.0f;
    v8f t;
#pragma unroll
    for (int r = 0; r < 8; ++r) t[r] = __builtin_amdgcn_rcpf(den[r]);
    v8f poly = ((((A5 * t + A4) * t + A3) * t + A2) * t + A1) * t;
    v8f m = (z * z) * -LOG2E;                                  // exp(-z^2)
    v8f e;
#pragma unroll
    for (int r = 0; r < 8; ++r) e[r] = __builtin_amdgcn_exp2f(m[r]);
    v8f q = (poly * e) * 0.5f;
    v8f res;
#pragma unroll
    for (int r = 0; r < 8; ++r) {
        float phi = (x[r] >= 0.0f) ? (1.0f - q[r]) : q[r];
        res[r] = x[r] * phi;
    }
    return res;
}
#endif

// A fragment (16x32 f16, M=samples): per ISA layout — lanes 0-15: M=lane,
// VGPR v<4 -> K=2v,2v+1 ; v>=4 -> K=16+2(v-4).. ; lanes 16-31 add K+=8.
// X is LDS [sample][feature] f16, strideH halves (even).
__device__ __forceinline__ v16h load_afrag(const _Float16* X, int strideH,
                                            int kb, int lane) {
    v8i a;
    const unsigned int* row = (const unsigned int*)(X + (lane & 15) * strideH);
    const int khi = (lane & 16) ? 8 : 0;
#pragma unroll
    for (int v = 0; v < 8; ++v) {
        int kp = kb + khi + 2 * (v & 3) + ((v & 4) ? 16 : 0);
        a[v] = (int)row[kp >> 1];
    }
    return __builtin_bit_cast(v16h, a);
}

// B fragment (32x16 f16, N=out-features): lane n = lane&15, elements 0-7 K=0..7,
// 8-15 K=16..23; lanes 16-31 add K+=8.  W is LDS [o][k] row-major f16.
__device__ __forceinline__ v16h load_bfrag(const unsigned short* W, int strideH,
                                            int obase, int kb, int lane) {
    int o = obase + (lane & 15);
    int k = kb + ((lane & 16) ? 8 : 0);
    const uint4* p = (const uint4*)(W + o * strideH + k);
    uint4 lo = p[0], hi = p[2];
    v8i a;
    a[0] = (int)lo.x; a[1] = (int)lo.y; a[2] = (int)lo.z; a[3] = (int)lo.w;
    a[4] = (int)hi.x; a[5] = (int)hi.y; a[6] = (int)hi.z; a[7] = (int)hi.w;
    return __builtin_bit_cast(v16h, a);
}

// ---------------------------------------------------------------------------
// Prep kernel: build InvConv weight from LU factors, convert weights to f16,
// init logdet[b] = G * (sum log|scale| + sum w_s).
// d_ws half layout: Wc[64*64] @0, W1[256*96] @4096, W2[64*256] @28672.
// ---------------------------------------------------------------------------
__global__ __launch_bounds__(256) void flow_prep(
    const float* __restrict__ w_p, const float* __restrict__ w_l,
    const float* __restrict__ w_s, const float* __restrict__ w_u,
    const float* __restrict__ s_sign, const float* __restrict__ scale,
    const float* __restrict__ W1, const float* __restrict__ W2,
    unsigned short* __restrict__ wts, float* __restrict__ logdet_out)
{
    __shared__ float Lp[4096];
    __shared__ float Up[4096];
    __shared__ float M1[4096];
    __shared__ float baseSh;
    const int t = threadIdx.x;

    for (int idx = t; idx < 4096; idx += 256) {
        int i = idx >> 6, j = idx & 63;
        Lp[idx] = (j < i) ? w_l[idx] : ((i == j) ? 1.0f : 0.0f);
        Up[idx] = (j > i) ? w_u[idx] : ((i == j) ? s_sign[i] * fast_exp(w_s[i]) : 0.0f);
    }
    __syncthreads();
    for (int idx = t; idx < 4096; idx += 256) {
        int i = idx >> 6, j = idx & 63;
        float acc = 0.0f;
        for (int k = 0; k < 64; ++k) acc += Lp[i * 64 + k] * Up[k * 64 + j];
        M1[idx] = acc;
    }
    __syncthreads();
    for (int idx = t; idx < 4096; idx += 256) {
        int i = idx >> 6, j = idx & 63;
        float acc = 0.0f;
        for (int k = 0; k < 64; ++k) acc += w_p[i * 64 + k] * M1[k * 64 + j];
        wts[idx] = f2h(acc);                       // Wc [o][c]
    }
    for (int idx = t; idx < H_ * 96; idx += 256) wts[4096 + idx] = f2h(W1[idx]);
    for (int idx = t; idx < C_ * H_; idx += 256) wts[28672 + idx] = f2h(W2[idx]);

    if (t == 0) {
        float s = 0.0f;
        for (int c = 0; c < 64; ++c) s += logf(fabsf(scale[c])) + w_s[c];
        baseSh = (float)G_ * s;
    }
    __syncthreads();
    logdet_out[t] = baseSh;                        // t == b (B_ == 256)
}

// ---------------------------------------------------------------------------
// Main fused kernel. 256 threads = 8 waves; each wave: 4 tiles of 16 g's.
// Dynamic LDS layout (bytes):
//   [0)       Wc f16 4096h, W1 f16 24576h, W2 f16 16384h      (90112 B)
//   [90112)   per-wave: X1 16x104h (3328 B) + X2 16x256h (8192 B)  x8
//   [182272)  b1 f32[256], b2 f32[64], scale f32[64], loc*scale f32[64]
// Total: 184064 B.
// ---------------------------------------------------------------------------
#define SMEM_BYTES 184064

__global__ __launch_bounds__(256) void flow_main(
    const float* __restrict__ input, const float* __restrict__ his,
    const float* __restrict__ loc,   const float* __restrict__ scale,
    const float* __restrict__ b1,    const float* __restrict__ b2,
    const unsigned short* __restrict__ wts,
    float* __restrict__ out, float* __restrict__ logdet)
{
    extern __shared__ unsigned char smem[];
    unsigned short* sW   = (unsigned short*)smem;           // Wc@0, W1@4096, W2@28672 (halves)
    _Float16* sWv        = (_Float16*)(sW + 45056);         // per-wave regions
    float* sF            = (float*)(smem + 182272);
    float* sB1 = sF;  float* sB2 = sF + 256;
    float* sSc = sF + 320; float* sLS = sF + 384;           // scale, loc*scale

    const int t = threadIdx.x, lane = t & 31, wave = t >> 5;
    const int m15 = lane & 15;
    const int hi8 = (lane & 16) ? 8 : 0;

    // stage weights + params into LDS
    {
        const unsigned int* src = (const unsigned int*)wts;
        unsigned int* dst = (unsigned int*)sW;
        for (int i = t; i < 22528; i += 256) dst[i] = src[i];
        sB1[t] = b1[t];
        if (t < 64) {
            float sc = scale[t];
            sB2[t] = b2[t]; sSc[t] = sc; sLS[t] = loc[t] * sc;
        }
    }
    __syncthreads();

    _Float16* X1 = sWv + wave * (1664 + 4096);              // [16][104] f16
    _Float16* X2 = X1 + 1664;                               // [16][256] f16
    _Float16* Y  = X2;                                      // alias [16][72] f16
    float*    Xo = (float*)X2;                              // alias [64][20] f32

    const unsigned short* Wc  = sW;
    const unsigned short* W1s = sW + 4096;
    const unsigned short* W2s = sW + 28672;

    const int waveGlobal = blockIdx.x * 8 + wave;

    for (int it = 0; it < 4; ++it) {
        const int tile = waveGlobal * 4 + it;               // 32768 tiles total
        const int bb = tile >> 7;                           // 128 tiles per batch
        const int g0 = (tile & 127) << 4;
        const size_t baseIn = (size_t)bb * (C_ * G_) + g0;

        // ---- Phase A: load input tile, ActNorm (x*sc + loc*sc) -> Y[s][c] ----
        {
            int c = lane >> 2;
            const int s0 = (lane & 3) * 4;
#pragma unroll
            for (int ic = 0; ic < 8; ++ic, c += 8) {
                float4 v = *(const float4*)(input + baseIn + (size_t)c * G_ + s0);
                float sc = sSc[c], ls = sLS[c];
                v2h p01 = pk2(v.x * sc + ls, v.y * sc + ls);
                v2h p23 = pk2(v.z * sc + ls, v.w * sc + ls);
                Y[(s0 + 0) * 72 + c] = p01.x;
                Y[(s0 + 1) * 72 + c] = p01.y;
                Y[(s0 + 2) * 72 + c] = p23.x;
                Y[(s0 + 3) * 72 + c] = p23.y;
            }
        }
        lds_fence_wave();

        // ---- Phase B: InvConv  z = Wc @ y   (D[m=sample][n=out-ch]) ----
        v16h ya0 = load_afrag(Y, 72, 0,  lane);
        v16h ya1 = load_afrag(Y, 72, 32, lane);
        v8f zacc[4];
#pragma unroll
        for (int nt = 0; nt < 4; ++nt) {
            v8f acc = {};
            acc = wmma_f16(ya0, load_bfrag(Wc, 64, nt * 16, 0,  lane), acc);
            acc = wmma_f16(ya1, load_bfrag(Wc, 64, nt * 16, 32, lane), acc);
            zacc[nt] = acc;
        }
        // in_b (ch 32..63) -> X1 features 64..95 (f16)
#pragma unroll
        for (int nt = 2; nt < 4; ++nt) {
            const int col = 64 + (nt - 2) * 16 + m15;
#pragma unroll
            for (int rp = 0; rp < 4; ++rp) {
                v2h p = pk2(zacc[nt][2 * rp], zacc[nt][2 * rp + 1]);
                X1[(2 * rp + 0 + hi8) * 104 + col] = p.x;
                X1[(2 * rp + 1 + hi8) * 104 + col] = p.y;
            }
        }

        // ---- Phase C: his_enc tile -> X1 features 0..63 ----
        {
            int c = lane >> 2;
            const int s0 = (lane & 3) * 4;
#pragma unroll
            for (int ic = 0; ic < 8; ++ic, c += 8) {
                float4 v = *(const float4*)(his + baseIn + (size_t)c * G_ + s0);
                v2h p01 = pk2(v.x, v.y);
                v2h p23 = pk2(v.z, v.w);
                X1[(s0 + 0) * 104 + c] = p01.x;
                X1[(s0 + 1) * 104 + c] = p01.y;
                X1[(s0 + 2) * 104 + c] = p23.x;
                X1[(s0 + 3) * 104 + c] = p23.y;
            }
        }
        lds_fence_wave();

        // ---- Phase D: MLP1 (96 -> 256) + bias + GELU -> X2[s][h] f16 ----
        v16h xa0 = load_afrag(X1, 104, 0,  lane);
        v16h xa1 = load_afrag(X1, 104, 32, lane);
        v16h xa2 = load_afrag(X1, 104, 64, lane);
#pragma unroll
        for (int nt = 0; nt < 16; ++nt) {
            v8f acc = {};
            acc = wmma_f16(xa0, load_bfrag(W1s, 96, nt * 16, 0,  lane), acc);
            acc = wmma_f16(xa1, load_bfrag(W1s, 96, nt * 16, 32, lane), acc);
            acc = wmma_f16(xa2, load_bfrag(W1s, 96, nt * 16, 64, lane), acc);
            const float bv = sB1[nt * 16 + m15];
            v8f g = gelu8(acc + bv);                       // packed add + GELU
            const int col = nt * 16 + m15;
#pragma unroll
            for (int rp = 0; rp < 4; ++rp) {
                v2h p = pk2(g[2 * rp], g[2 * rp + 1]);
                X2[(2 * rp + 0 + hi8) * 256 + col] = p.x;
                X2[(2 * rp + 1 + hi8) * 256 + col] = p.y;
            }
        }
        lds_fence_wave();

        // ---- Phase E: MLP2 (256 -> 64) + sigmoid, affine combine ----
        v16h ha[8];
#pragma unroll
        for (int kt = 0; kt < 8; ++kt) ha[kt] = load_afrag(X2, 256, kt * 32, lane);
        lds_fence_wave();                                   // X2 dead -> Xo reuse

        // in_b (f32) -> Xo cols 32..63
#pragma unroll
        for (int nt = 2; nt < 4; ++nt)
#pragma unroll
            for (int r = 0; r < 8; ++r)
                Xo[(nt * 16 + m15) * 20 + (r + hi8)] = zacc[nt][r];

        float lsum = 0.0f;
#pragma unroll
        for (int half = 0; half < 2; ++half) {              // (log_s nt, t nt+2)
            v8f accA = {}, accT = {};
#pragma unroll
            for (int kt = 0; kt < 8; ++kt)
                accA = wmma_f16(ha[kt], load_bfrag(W2s, 256, half * 16, kt * 32, lane), accA);
#pragma unroll
            for (int kt = 0; kt < 8; ++kt)
                accT = wmma_f16(ha[kt], load_bfrag(W2s, 256, (half + 2) * 16, kt * 32, lane), accT);
            const float hbA = 0.5f * sB2[half * 16 + m15];
            const float hbT = 0.5f * sB2[(half + 2) * 16 + m15];
#if HAVE_TANH
#pragma unroll
            for (int r = 0; r < 8; ++r) {
                // log_s = sigmoid(accA+b) = 0.5*tanh(0.5*(accA+b)) + 0.5
                float thA = __builtin_amdgcn_tanhf(accA[r] * 0.5f + hbA);
                float ls  = thA * 0.5f + 0.5f;
                // exp(log_s) = exp2(thA*0.5*log2e + 0.5*log2e)
                float ex  = __builtin_amdgcn_exp2f(thA * HLOG2E + HLOG2E);
                float tt  = __builtin_amdgcn_tanhf(accT[r] * 0.5f + hbT) * 0.5f + 0.5f;
                lsum += ls;
                Xo[(half * 16 + m15) * 20 + (r + hi8)] = ex * zacc[half][r] + tt;
            }
#else
#pragma unroll
            for (int r = 0; r < 8; ++r) {
                float ls = fast_sigmoid(accA[r] + 2.0f * hbA);
                float tt = fast_sigmoid(accT[r] + 2.0f * hbT);
                lsum += ls;
                Xo[(half * 16 + m15) * 20 + (r + hi8)] =
                    fast_exp(ls) * zacc[half][r] + tt;
            }
#endif
        }
        lds_fence_wave();

        // ---- Phase F: coalesced global store of out tile ----
        {
            int c = lane >> 2;
            const int s0 = (lane & 3) * 4;
#pragma unroll
            for (int ic = 0; ic < 8; ++ic, c += 8) {
                float4 v = *(const float4*)(Xo + c * 20 + s0);
                *(float4*)(out + baseIn + (size_t)c * G_ + s0) = v;
            }
        }

        // ---- logdet: wave reduce + atomic per batch ----
#pragma unroll
        for (int off = 16; off; off >>= 1) lsum += __shfl_xor(lsum, off, 32);
        if (lane == 0) unsafeAtomicAdd(logdet + bb, lsum);

        lds_fence_wave();                                   // Xo dead before next Y
    }
}

// ---------------------------------------------------------------------------
extern "C" void kernel_launch(void* const* d_in, const int* in_sizes, int n_in,
                              void* d_out, int out_size, void* d_ws, size_t ws_size,
                              hipStream_t stream) {
    const float* input  = (const float*)d_in[0];
    const float* his    = (const float*)d_in[1];
    const float* loc    = (const float*)d_in[2];
    const float* scale  = (const float*)d_in[3];
    const float* w_p    = (const float*)d_in[4];
    const float* w_l    = (const float*)d_in[5];
    const float* w_s    = (const float*)d_in[6];
    const float* w_u    = (const float*)d_in[7];
    const float* s_sign = (const float*)d_in[8];
    const float* W1     = (const float*)d_in[9];
    const float* b1     = (const float*)d_in[10];
    const float* W2     = (const float*)d_in[11];
    const float* b2     = (const float*)d_in[12];

    float* outp   = (float*)d_out;
    float* logdet = outp + (size_t)B_ * C_ * G_;
    unsigned short* wts = (unsigned short*)d_ws;

    (void)hipFuncSetAttribute((const void*)flow_main,
                              hipFuncAttributeMaxDynamicSharedMemorySize,
                              SMEM_BYTES);

    flow_prep<<<1, 256, 0, stream>>>(w_p, w_l, w_s, w_u, s_sign, scale,
                                     W1, W2, wts, logdet);
    flow_main<<<1024, 256, SMEM_BYTES, stream>>>(input, his, loc, scale,
                                                 b1, b2, wts, outp, logdet);
}
